// GraphConvDropoutBatch_80745385165392
// MI455X (gfx1250) — matrix operand order repaired and checked by reading, not compiled, not run on hardware
//
#include <hip/hip_runtime.h>

#define NN 50000
#define NE 800000
#define F  128
#define BN_EPS 1e-5f

typedef __attribute__((ext_vector_type(16))) _Float16 v16h;
typedef __attribute__((ext_vector_type(8)))  _Float16 v8h;
typedef __attribute__((ext_vector_type(8)))  float    v8f;

// ---------------- degree accumulation (edge-parallel, f32 atomics) ----------
__global__ void deg_kernel(const int* __restrict__ src, const int* __restrict__ dst,
                           float* __restrict__ deg_out, float* __restrict__ deg_in) {
  int e = blockIdx.x * blockDim.x + threadIdx.x;
  if (e < NE) {
    atomicAdd(&deg_out[src[e]], 1.0f);
    atomicAdd(&deg_in [dst[e]], 1.0f);
  }
}

// deg -> deg>0 ? rsqrt(deg) : 0, in place over both arrays (2N contiguous)
__global__ void invsqrt_kernel(float* __restrict__ d, int n) {
  int i = blockIdx.x * blockDim.x + threadIdx.x;
  if (i < n) {
    float v = d[i];
    d[i] = (v > 0.0f) ? rsqrtf(v) : 0.0f;
  }
}

// ---------------- SpMM: wave-per-edge gather + scatter-add ------------------
// 32 lanes x float4 = 128 contiguous floats per edge (coalesced 512B gather).
__global__ void scatter_kernel(const float* __restrict__ feat,
                               const int* __restrict__ src, const int* __restrict__ dst,
                               const float* __restrict__ inv_out,
                               float* __restrict__ agg) {
  int e    = blockIdx.x * 8 + (threadIdx.x >> 5);
  int lane = threadIdx.x & 31;
  if (e >= NE) return;
  int s = src[e];
  int d = dst[e];
  float w = inv_out[s];
  float4 v = ((const float4*)(feat + (size_t)s * F))[lane];
  float* ap = agg + (size_t)d * F + lane * 4;
  atomicAdd(ap + 0, v.x * w);
  atomicAdd(ap + 1, v.y * w);
  atomicAdd(ap + 2, v.z * w);
  atomicAdd(ap + 3, v.w * w);
}

// ---------------- WMMA GEMM + bias + dropout + BN-stat epilogue -------------
// Block: 256 threads = 8 waves. Tile: 64 rows x 128 cols. Each wave owns one
// 16-col N-tile and loops 4 M-tiles x 4 K-steps (K=32) => 16 v_wmma per wave.
__launch_bounds__(256)
__global__ void gemm_kernel(const float* __restrict__ agg,
                            const float* __restrict__ inv_in,
                            const float* __restrict__ W,
                            const float* __restrict__ bias,
                            const float* __restrict__ mask,
                            float* __restrict__ out,
                            float* __restrict__ col_sum,
                            float* __restrict__ col_sumsq) {
  __shared__ __align__(32) _Float16 sA[64][136];   // padded rows: bank skew
  __shared__ __align__(32) _Float16 sB[16384];     // W pre-swizzled to B layout

  const int tid = threadIdx.x;
  const int m0  = blockIdx.x * 64;

  // --- stage A: 64 x 128 f32 -> f16, scaled by inv_sqrt_in[row] -------------
  #pragma unroll
  for (int j = 0; j < 8; ++j) {
    int idx = tid + j * 256;            // float4 index, 0..2047
    int row = idx >> 5;                 // 32 float4 per row
    int c4  = idx & 31;
    int gr  = m0 + row;
    float4 v = make_float4(0.f, 0.f, 0.f, 0.f);
    if (gr < NN) {
      v = ((const float4*)(agg + (size_t)gr * F))[c4];
      float w = inv_in[gr];
      v.x *= w; v.y *= w; v.z *= w; v.w *= w;
    }
    int k = c4 * 4;
    sA[row][k + 0] = (_Float16)v.x;
    sA[row][k + 1] = (_Float16)v.y;
    sA[row][k + 2] = (_Float16)v.z;
    sA[row][k + 3] = (_Float16)v.w;
  }

  // --- stage B: swizzle W[k][n] into per-lane WMMA B layout -----------------
  // 32x16 f16 B tile, contiguous k order used here (same as A): lanes 0-15 -> N=lane, halves = K 0..15 of this K-step;
  // lanes 16-31 -> N=lane-16, halves = K 16..31.
  #pragma unroll
  for (int j = 0; j < 64; ++j) {
    int idx = tid + j * 256;            // coalesced: idx = k*128 + n
    int k = idx >> 7;
    int n = idx & 127;
    _Float16 w = (_Float16)W[idx];
    int kstep = k >> 5, kr = k & 31;
    int ntile = n >> 4, nl = n & 15;
    int lane  = nl + ((kr >> 4) << 4);
    int i     = kr & 15;
    sB[(((kstep * 8 + ntile) * 32) + lane) * 16 + i] = w;
  }
  __syncthreads();

  const int lane = tid & 31;
  const int wave = tid >> 5;            // == ntile (N-tile index)
  const int hs   = lane >> 4;           // half-wave select
  const int ml   = lane & 15;

  v8f acc[4] = {};                      // 4 M-tiles of 16x16 f32 accumulators

  #pragma unroll
  for (int kstep = 0; kstep < 4; ++kstep) {
    v16h b = *(const v16h*)&sB[(((kstep * 8 + wave) * 32) + lane) * 16];
    #pragma unroll
    for (int mt = 0; mt < 4; ++mt) {
      // A 16x32 f16, contiguous k order used here (same as B): lane<16 -> M=lane, K 0-15; lane>=16 -> K 16-31
      const _Float16* ap = &sA[(mt << 4) + ml][(kstep << 5) + (hs << 4)];
      v8h lo = *(const v8h*)(ap);
      v8h hi = *(const v8h*)(ap + 8);
      v16h a;
      #pragma unroll
      for (int i = 0; i < 8; ++i) { a[i] = lo[i]; a[i + 8] = hi[i]; }
      acc[mt] = __builtin_amdgcn_wmma_f32_16x16x32_f16(
          false, a, false, b, (short)0, acc[mt], false, false);
    }
  }

  // --- epilogue: bias + dropout(mask * 1/(1-p)) + write + BN stats ----------
  const int col  = (wave << 4) + ml;    // this lane's output column
  const float bc = bias[col];
  float lsum = 0.f, lsq = 0.f;
  #pragma unroll
  for (int mt = 0; mt < 4; ++mt) {
    #pragma unroll
    for (int r = 0; r < 8; ++r) {
      int row = m0 + (mt << 4) + (hs << 3) + r;   // C/D: VGPR r -> M=r (+8 for hi half)
      if (row < NN) {
        float val = acc[mt][r] + bc;
        val = val * mask[(size_t)row * F + col] * 2.0f;   // 1/(1-0.5)
        out[(size_t)row * F + col] = val;
        lsum += val;
        lsq  += val * val;
      }
    }
  }
  atomicAdd(&col_sum  [col], lsum);
  atomicAdd(&col_sumsq[col], lsq);
}

// ---------------- BN: finalize per-column scale/shift -----------------------
__global__ void bnstats_kernel(const float* __restrict__ col_sum,
                               const float* __restrict__ col_sumsq,
                               const float* __restrict__ gamma,
                               const float* __restrict__ beta,
                               float* __restrict__ scale,
                               float* __restrict__ shift) {
  int c = threadIdx.x;
  if (c < F) {
    float invN = 1.0f / (float)NN;
    float mean = col_sum[c] * invN;
    float var  = col_sumsq[c] * invN - mean * mean;
    float inv  = rsqrtf(var + BN_EPS);
    float sc   = gamma[c] * inv;
    scale[c] = sc;
    shift[c] = beta[c] - mean * sc;
  }
}

// ---------------- BN: apply (vectorized float4) -----------------------------
__global__ void bnapply_kernel(float* __restrict__ out,
                               const float* __restrict__ scale,
                               const float* __restrict__ shift) {
  int i4 = blockIdx.x * blockDim.x + threadIdx.x;
  if (i4 < (NN * F) / 4) {
    float4 v = ((float4*)out)[i4];
    int c = (i4 & 31) * 4;
    v.x = v.x * scale[c + 0] + shift[c + 0];
    v.y = v.y * scale[c + 1] + shift[c + 1];
    v.z = v.z * scale[c + 2] + shift[c + 2];
    v.w = v.w * scale[c + 3] + shift[c + 3];
    ((float4*)out)[i4] = v;
  }
}

extern "C" void kernel_launch(void* const* d_in, const int* in_sizes, int n_in,
                              void* d_out, int out_size, void* d_ws, size_t ws_size,
                              hipStream_t stream) {
  const float* feat  = (const float*)d_in[0];
  const int*   src   = (const int*)  d_in[1];
  const int*   dst   = (const int*)  d_in[2];
  const float* W     = (const float*)d_in[3];
  const float* b     = (const float*)d_in[4];
  const float* gamma = (const float*)d_in[5];
  const float* beta  = (const float*)d_in[6];
  const float* mask  = (const float*)d_in[7];
  float* out = (float*)d_out;

  // workspace layout (floats): [deg/inv_out N][deg/inv_in N][agg N*F][sum 128][sq 128][scale 128][shift 128]
  float* ws        = (float*)d_ws;
  float* inv_out   = ws;
  float* inv_in    = ws + NN;
  float* agg       = ws + 2 * NN;
  float* col_sum   = agg + (size_t)NN * F;
  float* col_sumsq = col_sum + F;
  float* scale     = col_sumsq + F;
  float* shift     = scale + F;

  hipMemsetAsync(inv_out, 0, sizeof(float) * 2 * NN, stream);
  hipMemsetAsync(agg,     0, sizeof(float) * (size_t)NN * F, stream);
  hipMemsetAsync(col_sum, 0, sizeof(float) * 2 * F, stream);

  deg_kernel    <<<(NE + 255) / 256,      256, 0, stream>>>(src, dst, inv_out, inv_in);
  invsqrt_kernel<<<(2 * NN + 255) / 256,  256, 0, stream>>>(inv_out, 2 * NN);
  scatter_kernel<<<NE / 8,                256, 0, stream>>>(feat, src, dst, inv_out, agg);
  gemm_kernel   <<<(NN + 63) / 64,        256, 0, stream>>>(agg, inv_in, W, b, mask, out,
                                                            col_sum, col_sumsq);
  bnstats_kernel<<<1,                     128, 0, stream>>>(col_sum, col_sumsq, gamma, beta,
                                                            scale, shift);
  bnapply_kernel<<<(NN * F / 4 + 255) / 256, 256, 0, stream>>>(out, scale, shift);
}
